// CA_26431228740146
// MI455X (gfx1250) — compile-verified, matrix operand order checked by reading, not run on hardware
//
#include <hip/hip_runtime.h>

typedef _Float16 half16 __attribute__((ext_vector_type(16)));
typedef _Float16 half8  __attribute__((ext_vector_type(8)));
typedef float    f32x8  __attribute__((ext_vector_type(8)));
typedef float    f32x4  __attribute__((ext_vector_type(4)));

#define BATCH 32
#define HDIM  256
#define WDIM  256
#define CH    16
#define TW    16
#define TH    8
#define K1P   64    // GEMM1 K padded 48 -> 64
#define DH    128   // hidden dim

// Load a 16x32 fp16 WMMA A/B fragment chunk from LDS.
// p points at [row][chunk_base]; lane-dependent basek = (lane&16)?8:0.
// Per CDNA5 ISA 7.12.2: VGPR0..3 = K[basek..basek+7], VGPR4..7 = K[basek+16..basek+23].
__device__ __forceinline__ half16 ldsFrag(const _Float16* p, int basek) {
  half8 lo = *(const half8*)(p + basek);
  half8 hi = *(const half8*)(p + basek + 16);
  half16 r;
#pragma unroll
  for (int i = 0; i < 8; ++i) { r[i] = lo[i]; r[i + 8] = hi[i]; }
  return r;
}

// ---------------- Kernel A: async halo tile -> perceive + 2-layer MLP (WMMA) + update ----------------
__global__ __launch_bounds__(128) void nca_update(
    const float* __restrict__ x, const float* __restrict__ rmask,
    const float* __restrict__ W1, const float* __restrict__ B1,
    const float* __restrict__ W2, const float* __restrict__ B2,
    float* __restrict__ xnew)
{
  __shared__ __align__(16) _Float16 s_y[4][32][K1P];     // 16 KB  perceive features per wave
  // s_hx aliases: phase 1 = fp32 halo tile s_x[10][18][16] (11.5 KB, dead after barrier 2)
  //               phase 2 = per-wave hidden acts s_h[4][16][128] fp16 (16 KB)
  __shared__ __align__(16) unsigned char s_hx[4 * 16 * DH * 2];  // 16 KB
  __shared__ __align__(16) _Float16 s_w1[DH][K1P];       // 16 KB  W1^T fp16, K padded
  __shared__ __align__(16) _Float16 s_w2[CH][DH];        //  4 KB  W2^T fp16
  __shared__ float s_b1[DH];
  __shared__ float s_b2[CH];

  float    (*s_x)[18][CH]  = (float (*)[18][CH])s_hx;          // [10][18][16] fp32 halo
  _Float16 (*s_h)[16][DH]  = (_Float16 (*)[16][DH])s_hx;       // [4][16][128] fp16 hidden

  const int t    = threadIdx.x;
  const int lane = t & 31;
  const int wave = t >> 5;
  const int x0 = blockIdx.x * TW;
  const int y0 = blockIdx.y * TH;
  const int bz = blockIdx.z;

  // ---- phase 0: async-DMA the 18x10 pixel halo tile (fp32, 16ch) into LDS ----
  // 180 halo pixels * 4 x b128 = 720 async transfers over 128 threads.
#pragma unroll 1
  for (int i = t; i < 720; i += 128) {
    int pix  = i >> 2;
    int part = i & 3;
    int hy = pix / 18;           // 0..9
    int hx = pix - hy * 18;      // 0..17
    int gy2 = y0 + hy - 1;
    int gx2 = x0 + hx - 1;
    float* dst = &s_x[hy][hx][part * 4];
    if (gy2 >= 0 && gy2 < HDIM && gx2 >= 0 && gx2 < WDIM) {
      unsigned ldsa = (unsigned)(size_t)dst;   // low 32 bits of generic addr = LDS offset
      int goff = (int)(((((bz * HDIM + gy2) * WDIM) + gx2) * CH + part * 4) * 4); // bytes
      asm volatile("global_load_async_to_lds_b128 %0, %1, %2"
                   :: "v"(ldsa), "v"(goff), "s"(x)
                   : "memory");
    } else {
      *(f32x4*)dst = (f32x4){0.f, 0.f, 0.f, 0.f};   // zero pad (never beats 0.1 thresh / conv 0)
    }
  }

  // ---- stage weights fp32 -> fp16 into LDS while the async DMA flies ----
#pragma unroll 1
  for (int i = t; i < DH * K1P; i += 128) {
    int n = i & (DH - 1);
    int k = i >> 7;
    s_w1[n][k] = (k < 48) ? (_Float16)W1[k * DH + n] : (_Float16)0.0f;
  }
#pragma unroll 1
  for (int i = t; i < CH * DH; i += 128) {
    int n = i & (CH - 1);
    int k = i >> 4;
    s_w2[n][k] = (_Float16)W2[k * CH + n];
  }
  if (t < DH) s_b1[t] = B1[t];
  if (t < CH) s_b2[t] = B2[t];

  asm volatile("s_wait_asynccnt 0x0" ::: "memory");   // async tile resident in LDS
  __syncthreads();                                     // barrier 1: tile + weights visible

  // ---- perceive: one pixel per thread, neighborhoods from LDS ----
  const int px = t & 15, py = t >> 4;
  const int pw = lane;               // pixel slot in this wave's region

#pragma unroll
  for (int k = 48; k < K1P; ++k) s_y[wave][pw][k] = (_Float16)0.0f;

#pragma unroll 1
  for (int cg = 0; cg < 4; ++cg) {   // 4 channels per pass
    f32x4 nb[3][3];
#pragma unroll
    for (int dy = 0; dy < 3; ++dy)
#pragma unroll
      for (int dxk = 0; dxk < 3; ++dxk)
        nb[dy][dxk] = *(const f32x4*)&s_x[py + dy][px + dxk][cg * 4];
#pragma unroll
    for (int i = 0; i < 4; ++i) {
      // effective kernels from the reference's (filter,row)->HW reshape:
      // K0 = [[-1,-2,-1],[-1,0,1],[0,0,0]]/8 ; K1 = [[0,0,0],[-2,0,2],[0,8,0]]/8 ;
      // K2 = [[ 1, 2, 1],[-1,0,1],[0,0,0]]/8
      float a00 = nb[0][0][i], a01 = nb[0][1][i], a02 = nb[0][2][i];
      float a10 = nb[1][0][i], a12 = nb[1][2][i];
      float a21 = nb[2][1][i];
      float yj0 = (-a00 - 2.f * a01 - a02 - a10 + a12) * 0.125f;
      float yj1 = (a12 - a10) * 0.25f + a21;
      float yj2 = ( a00 + 2.f * a01 + a02 - a10 + a12) * 0.125f;
      int ch = cg * 4 + i;
      s_y[wave][pw][ch * 3 + 0] = (_Float16)yj0;
      s_y[wave][pw][ch * 3 + 1] = (_Float16)yj1;
      s_y[wave][pw][ch * 3 + 2] = (_Float16)yj2;
    }
  }

  __syncthreads();   // barrier 2: all s_x reads done -> s_hx may be reused as s_h

  // ---- MLP via WMMA: each wave owns 2 M-tiles (its own 32 pixels) ----
  const int basek  = (lane & 16) ? 8 : 0;
  const int rsel   = lane & 15;
  const int rowoff = (lane & 16) ? 8 : 0;

#pragma unroll 1
  for (int sub = 0; sub < 2; ++sub) {
    const _Float16* yrow = &s_y[wave][sub * 16 + rsel][0];
    half16 a0 = ldsFrag(yrow, basek);       // K 0..31
    half16 a1 = ldsFrag(yrow + 32, basek);  // K 32..63

    // GEMM1: [16x64] x [64x128] -> relu -> s_h
#pragma unroll
    for (int nt = 0; nt < 8; ++nt) {
      const _Float16* wrow = &s_w1[nt * 16 + rsel][0];
      half16 b0  = ldsFrag(wrow, basek);
      half16 b1v = ldsFrag(wrow + 32, basek);
      f32x8 acc = {};
      acc = __builtin_amdgcn_wmma_f32_16x16x32_f16(false, a0, false, b0,  (short)0, acc, false, false);
      acc = __builtin_amdgcn_wmma_f32_16x16x32_f16(false, a1, false, b1v, (short)0, acc, false, false);
      float bias = s_b1[nt * 16 + rsel];
#pragma unroll
      for (int r = 0; r < 8; ++r) {
        float hv = fmaxf(acc[r] + bias, 0.0f);
        s_h[wave][r + rowoff][nt * 16 + rsel] = (_Float16)hv;   // wave-local; DS in-order
      }
    }

    // GEMM2: [16x128] x [128x16]
    f32x8 acc2 = {};
#pragma unroll
    for (int kk = 0; kk < 4; ++kk) {
      half16 ah = ldsFrag(&s_h[wave][rsel][kk * 32], basek);
      half16 bh = ldsFrag(&s_w2[rsel][kk * 32], basek);
      acc2 = __builtin_amdgcn_wmma_f32_16x16x32_f16(false, ah, false, bh, (short)0, acc2, false, false);
    }

    // stochastic update: x_new = x + dx * (rand < 0.5)
    const int   chan  = rsel;
    const float bias2 = s_b2[chan];
#pragma unroll
    for (int r = 0; r < 8; ++r) {
      int p   = wave * 32 + sub * 16 + rowoff + r;   // block-local pixel
      int ppy = p >> 4, ppx = p & 15;
      size_t pidx = (size_t)(bz * HDIM + (y0 + ppy)) * WDIM + (x0 + ppx);
      float upd = (rmask[pidx] < 0.5f) ? 1.0f : 0.0f;
      size_t idx = pidx * CH + chan;
      float dxv = acc2[r] + bias2;
      xnew[idx] = x[idx] + dxv * upd;                // lanes 0-15 / 16-31 -> 2x 64B coalesced
    }
  }
}

// ---------------- Kernel B: pre/post alive masks + final store ----------------
__global__ __launch_bounds__(256) void nca_mask(
    const float* __restrict__ x, const float* __restrict__ xnew, float* __restrict__ out)
{
  __shared__ float s_pre[18][18];
  __shared__ float s_post[18][18];
  const int t  = threadIdx.x;
  const int x0 = blockIdx.x * 16, y0 = blockIdx.y * 16, bz = blockIdx.z;

#pragma unroll 1
  for (int i = t; i < 18 * 18; i += 256) {
    int ly = i / 18, lx = i - ly * 18;
    int gy = y0 + ly - 1, gx = x0 + lx - 1;
    float pa = 0.0f, qa = 0.0f;                      // OOB=0: never exceeds 0.1 threshold
    if (gy >= 0 && gy < HDIM && gx >= 0 && gx < WDIM) {
      size_t idx = ((size_t)(bz * HDIM + gy) * WDIM + gx) * CH + 3;  // alpha channel
      pa = x[idx];
      qa = xnew[idx];
    }
    s_pre[ly][lx]  = pa;
    s_post[ly][lx] = qa;
  }
  __syncthreads();

  const int px = t & 15, py = t >> 4;
  float mpre = -1e30f, mpost = -1e30f;
#pragma unroll
  for (int dy = 0; dy < 3; ++dy)
#pragma unroll
    for (int dxk = 0; dxk < 3; ++dxk) {
      mpre  = fmaxf(mpre,  s_pre[py + dy][px + dxk]);
      mpost = fmaxf(mpost, s_post[py + dy][px + dxk]);
    }
  float life = (mpre > 0.1f && mpost > 0.1f) ? 1.0f : 0.0f;

  size_t base = ((size_t)(bz * HDIM + (y0 + py)) * WDIM + (x0 + px)) * CH;
#pragma unroll
  for (int c = 0; c < CH; c += 4) {
    f32x4 v = *(const f32x4*)(xnew + base + c);
    v = v * life;
    *(f32x4*)(out + base + c) = v;
  }
}

extern "C" void kernel_launch(void* const* d_in, const int* in_sizes, int n_in,
                              void* d_out, int out_size, void* d_ws, size_t ws_size,
                              hipStream_t stream) {
  (void)in_sizes; (void)n_in; (void)out_size; (void)ws_size;
  const float* x  = (const float*)d_in[0];
  const float* rm = (const float*)d_in[1];
  const float* W1 = (const float*)d_in[2];
  const float* B1 = (const float*)d_in[3];
  const float* W2 = (const float*)d_in[4];
  const float* B2 = (const float*)d_in[5];
  float* out  = (float*)d_out;
  float* xnew = (float*)d_ws;   // needs 32*256*256*16*4 = 128 MiB scratch

  dim3 gA(WDIM / TW, HDIM / TH, BATCH);   // 16 x 32 x 32 blocks, 128 thr (4 waves)
  nca_update<<<gA, 128, 0, stream>>>(x, rm, W1, B1, W2, B2, xnew);

  dim3 gB(WDIM / 16, HDIM / 16, BATCH);   // 16 x 16 x 32 blocks, 256 thr
  nca_mask<<<gB, 256, 0, stream>>>(x, xnew, out);
}
